// Net_40501541601792
// MI455X (gfx1250) — compile-verified, matrix operand order checked by reading, not run on hardware
//
#include <hip/hip_runtime.h>
#include <hip/hip_bf16.h>
#include <cstdint>
#include <cfloat>

#define NKER 125
#define KSP  5

typedef __attribute__((ext_vector_type(16))) __bf16        v16bf;
typedef __attribute__((ext_vector_type(8)))  float         v8f;
typedef __attribute__((ext_vector_type(4)))  unsigned int  v4u;

// ---------------- small device utils ----------------
__device__ __forceinline__ unsigned short f32_bf16(float f) {
  unsigned u = __float_as_uint(f);
  unsigned r = (u + 0x7FFFu + ((u >> 16) & 1u)) >> 16;   // round-to-nearest-even
  return (unsigned short)r;
}

__device__ __forceinline__ void atomicMaxF(float* addr, float val) {
  unsigned* ua = (unsigned*)addr;
  unsigned cur = *ua;
  while (__uint_as_float(cur) < val) {
    unsigned prev = atomicCAS(ua, cur, __float_as_uint(val));
    if (prev == cur) break;
    cur = prev;
  }
}

// ---------------- generic fill ----------------
__global__ void fill_f32(float* __restrict__ p, float v, long long n) {
  long long i = (long long)blockIdx.x * blockDim.x + threadIdx.x;
  if (i < n) p[i] = v;
}

// ---------------- Cartesian pseudo-coords ----------------
__global__ void cart_max_kernel(const float* __restrict__ pos, const int* __restrict__ ei,
                                int E, float* __restrict__ cart, float* __restrict__ maxabs) {
  int e = blockIdx.x * blockDim.x + threadIdx.x;
  if (e >= E) return;
  int s = ei[e], d = ei[E + e];
  float mx = 0.f;
#pragma unroll
  for (int k = 0; k < 3; ++k) {
    float c = pos[s * 3 + k] - pos[d * 3 + k];
    cart[e * 3 + k] = c;
    mx = fmaxf(mx, fabsf(c));
  }
  atomicMax((unsigned*)maxabs, __float_as_uint(mx));   // bits monotone for >= 0
}

__global__ void pseudo_kernel(float* __restrict__ cart, const float* __restrict__ maxabs, long long n3) {
  long long i = (long long)blockIdx.x * blockDim.x + threadIdx.x;
  if (i >= n3) return;
  cart[i] = cart[i] / (2.0f * maxabs[0]) + 0.5f;
}

// ---------------- spline basis + scatter into buckets ----------------
// S row length is KP (K padded to multiple of 64); bucket (dst, wi) at
// S[dst*KP + wi*cin .. +cin). Pad region stays zero.
__global__ void spline_scatter(const float* __restrict__ x, const int* __restrict__ ei,
                               const float* __restrict__ pseudo,
                               float* __restrict__ S, float* __restrict__ deg,
                               int E, int cin, int KP) {
  long long t = (long long)blockIdx.x * blockDim.x + threadIdx.x;
  if (t >= (long long)E * 8) return;
  int e = (int)(t >> 3);
  int s = (int)(t & 7);
  int src = ei[e], dst = ei[E + e];
  float w = 1.f;
  int wi = 0, mul = 1;
#pragma unroll
  for (int d = 0; d < 3; ++d) {
    float v  = fminf(fmaxf(pseudo[e * 3 + d], 0.f), 1.f) * (float)(KSP - 1);
    float fl = floorf(v);
    float fr = v - fl;
    int bit = (s >> d) & 1;
    int id = (int)fl + bit;
    id = id < 0 ? 0 : (id > KSP - 1 ? KSP - 1 : id);
    wi += mul * id;
    mul *= KSP;
    w *= bit ? fr : (1.f - fr);
  }
  if (s == 0) atomicAdd(&deg[dst], 1.f);
  float* Srow = S + (long long)dst * KP + (long long)wi * cin;
  const float* xrow = x + (long long)src * cin;
  for (int i = 0; i < cin; ++i) atomicAdd(&Srow[i], w * xrow[i]);
}

// ---------------- f32 -> bf16 cast with zero row padding (weights) ----------------
__global__ void cast_bf16_pad(const float* __restrict__ in, unsigned short* __restrict__ out,
                              int KK, int KP, int cout) {
  long long t = (long long)blockIdx.x * blockDim.x + threadIdx.x;
  if (t >= (long long)KP * cout) return;
  int k = (int)(t / cout);
  out[t] = (k < KK) ? f32_bf16(in[t]) : (unsigned short)0;
}

// ---------------- WMMA GEMM: C[M,N] = A_f32[M,KP] * B_bf16[KP,N] ----------------
// 256 threads = 8 waves; block tile 32(M) x 64(N); K step 64 (2 WMMAs / barrier).
// KP % 64 == 0 -> no bounds checks anywhere. LDS tiles are packed-bf16 in
// fragment-native layout (K pairs contiguous; A m-major, B n-major/transposed;
// row stride 72 shorts = 144B, 16B-aligned) so one WMMA operand is exactly
// two ds_load_b128 per lane, with zero VALU in the fragment path.
__global__ __launch_bounds__(256)
void wmma_gemm(const float* __restrict__ A, const unsigned short* __restrict__ Bb,
               float* __restrict__ C, int M, int N, int KP) {
  __shared__ unsigned short As[32][72];   // [m][k] bf16
  __shared__ unsigned short Bs[64][72];   // [n][k] bf16 (transposed)
  const int m0 = blockIdx.x * 32;
  const int n0 = blockIdx.y * 64;
  const int tid  = threadIdx.x;
  const int wave = tid >> 5;
  const int lane = tid & 31;
  const int wm = (wave >> 2) * 16;   // 0 / 16
  const int wn = (wave & 3) * 16;    // 0..48
  const int kp0 = (lane >> 4) * 4;   // K-pair base within a 32-K block (ISA layout)
  v8f acc = {};

  for (int k0 = 0; k0 < KP; k0 += 64) {
    // stage A tile: 32 rows x 16 groups of 4 K; b128 load -> cvt -> b64 store
#pragma unroll
    for (int r = 0; r < 2; ++r) {
      int idx = tid + r * 256;
      int m = idx >> 4, kq = idx & 15;
      const float4 f = *(const float4*)(A + (long long)(m0 + m) * KP + k0 + 4 * kq);
      unsigned p0 = (unsigned)f32_bf16(f.x) | ((unsigned)f32_bf16(f.y) << 16);
      unsigned p1 = (unsigned)f32_bf16(f.z) | ((unsigned)f32_bf16(f.w) << 16);
      uint2 pk; pk.x = p0; pk.y = p1;
      *(uint2*)&As[m][4 * kq] = pk;
    }
    // stage B tile transposed: 32 K-pairs x 32 N-pairs; 2x b32 load -> 2x b32 store
#pragma unroll
    for (int r = 0; r < 4; ++r) {
      int idx = tid + r * 256;
      int kp = idx >> 5, np = idx & 31;
      int k = k0 + 2 * kp;
      unsigned ua = *(const unsigned*)(Bb + (long long)k * N + n0 + 2 * np);
      unsigned ub = *(const unsigned*)(Bb + (long long)(k + 1) * N + n0 + 2 * np);
      *(unsigned*)&Bs[2 * np][2 * kp]     = (ua & 0xFFFFu) | (ub << 16);
      *(unsigned*)&Bs[2 * np + 1][2 * kp] = (ua >> 16) | (ub & 0xFFFF0000u);
    }
    // prefetch next K tile into cache (global_prefetch_b8)
    if (k0 + 64 < KP) {
      __builtin_prefetch(&A[(long long)(m0 + (tid >> 3)) * KP + (k0 + 64)], 0, 0);
      __builtin_prefetch(&Bb[(long long)(k0 + 64 + (tid >> 2)) * N + n0], 0, 0);
    }
    __syncthreads();

    const unsigned short* arow = &As[wm + (lane & 15)][0];
    const unsigned short* brow = &Bs[wn + (lane & 15)][0];
    union Frag { v16bf v; v4u q[2]; };
#pragma unroll
    for (int half = 0; half < 2; ++half) {
      const int base = half * 16;            // K-pair offset of this 32-K block
      Frag af, bf_;
      af.q[0]  = *(const v4u*)&arow[2 * (base + kp0)];
      af.q[1]  = *(const v4u*)&arow[2 * (base + kp0 + 8)];
      bf_.q[0] = *(const v4u*)&brow[2 * (base + kp0)];
      bf_.q[1] = *(const v4u*)&brow[2 * (base + kp0 + 8)];
      acc = __builtin_amdgcn_wmma_f32_16x16x32_bf16(false, af.v, false, bf_.v,
                                                    (short)0, acc, false, false);
    }
    __syncthreads();
  }

  // C/D layout: VGPR r -> M = r + 8*(lane/16), N = lane%16
#pragma unroll
  for (int r = 0; r < 8; ++r) {
    int mm = m0 + wm + r + 8 * (lane >> 4);
    int nn = n0 + wn + (lane & 15);
    C[(long long)mm * N + nn] = acc[r];
  }
}

// ---------------- epilogue: mean + root GEMV + bias + ELU ----------------
__global__ void epilogue_kernel(const float* __restrict__ msg, const float* __restrict__ deg,
                                const float* __restrict__ x, const float* __restrict__ root,
                                const float* __restrict__ bias, float* __restrict__ out,
                                int Nn, int cin, int cout) {
  long long t = (long long)blockIdx.x * blockDim.x + threadIdx.x;
  if (t >= (long long)Nn * cout) return;
  int n = (int)(t / cout), o = (int)(t % cout);
  float r = 0.f;
  const float* xr = x + (long long)n * cin;
  for (int i = 0; i < cin; ++i) r += xr[i] * root[(long long)i * cout + o];
  float dv = deg[n];
  dv = dv > 1.f ? dv : 1.f;
  float v = msg[t] / dv + r + bias[o];
  out[t] = v > 0.f ? v : expm1f(v);
}

// ---------------- BatchNorm (train stats), in-place, block per channel ----------------
__global__ void bn_kernel(float* __restrict__ h, const float* __restrict__ g,
                          const float* __restrict__ be, int Nn, int cout) {
  int o = blockIdx.x;
  __shared__ float red[256];
  __shared__ float stats[2];
  float s = 0.f, s2 = 0.f;
  for (int n = threadIdx.x; n < Nn; n += 256) {
    float v = h[(long long)n * cout + o];
    s += v; s2 += v * v;
  }
  red[threadIdx.x] = s; __syncthreads();
  for (int st = 128; st > 0; st >>= 1) { if (threadIdx.x < st) red[threadIdx.x] += red[threadIdx.x + st]; __syncthreads(); }
  if (threadIdx.x == 0) stats[0] = red[0] / (float)Nn;
  __syncthreads();
  red[threadIdx.x] = s2; __syncthreads();
  for (int st = 128; st > 0; st >>= 1) { if (threadIdx.x < st) red[threadIdx.x] += red[threadIdx.x + st]; __syncthreads(); }
  if (threadIdx.x == 0) stats[1] = red[0] / (float)Nn - stats[0] * stats[0];
  __syncthreads();
  float mean = stats[0];
  float rstd = rsqrtf(stats[1] + 1e-5f);
  float gg = g[o], bb = be[o];
  for (int n = threadIdx.x; n < Nn; n += 256) {
    long long ix = (long long)n * cout + o;
    h[ix] = (h[ix] - mean) * rstd * gg + bb;
  }
}

// ---------------- max-pool scatter / finalize ----------------
__global__ void pool_scatter(const float* __restrict__ h, const float* __restrict__ pos,
                             const int* __restrict__ cl, float* __restrict__ xm,
                             float* __restrict__ cnt, float* __restrict__ psum,
                             int Nin, int cout, int hasPos) {
  int n = blockIdx.x * blockDim.x + threadIdx.x;
  if (n >= Nin) return;
  int c = cl[n];
  atomicAdd(&cnt[c], 1.f);
  if (hasPos) {
#pragma unroll
    for (int d = 0; d < 3; ++d) atomicAdd(&psum[c * 3 + d], pos[n * 3 + d]);
  }
  for (int o = 0; o < cout; ++o)
    atomicMaxF(&xm[(long long)c * cout + o], h[(long long)n * cout + o]);
}

__global__ void pool_fin_x(float* __restrict__ xm, const float* __restrict__ cnt,
                           long long n, int cout) {
  long long t = (long long)blockIdx.x * blockDim.x + threadIdx.x;
  if (t >= n) return;
  if (cnt[t / cout] <= 0.f) xm[t] = 0.f;
}

__global__ void pool_fin_pos(const float* __restrict__ psum, const float* __restrict__ cnt,
                             float* __restrict__ pout, int nout) {
  int t = blockIdx.x * blockDim.x + threadIdx.x;
  if (t >= nout * 3) return;
  float c = cnt[t / 3];
  pout[t] = psum[t] / (c > 1.f ? c : 1.f);
}

// ---------------- FC head ----------------
__global__ void fc1_kernel(const float* __restrict__ hc, const float* __restrict__ W,
                           const float* __restrict__ b, float* __restrict__ z,
                           int In, int On) {
  int t = blockIdx.x * blockDim.x + threadIdx.x;
  if (t >= 8 * On) return;
  int bb = t / On, o = t % On;
  float s = b[o];
  const float* hr = hc + (long long)bb * In;
  for (int i = 0; i < In; ++i) s += hr[i] * W[(long long)i * On + o];
  z[t] = s > 0.f ? s : expm1f(s);
}

__global__ void fc2_lsm(const float* __restrict__ z, const float* __restrict__ W,
                        const float* __restrict__ b, float* __restrict__ out) {
  __shared__ float logits[8][10];
  int t = threadIdx.x;
  if (t < 80) {
    int bb = t / 10, o = t % 10;
    float s = b[o];
    for (int i = 0; i < 1024; ++i) s += z[bb * 1024 + i] * W[i * 10 + o];
    logits[bb][o] = s;
  }
  __syncthreads();
  if (t < 80) {
    int bb = t / 10, o = t % 10;
    float mx = -FLT_MAX;
    for (int j = 0; j < 10; ++j) mx = fmaxf(mx, logits[bb][j]);
    float se = 0.f;
    for (int j = 0; j < 10; ++j) se += expf(logits[bb][j] - mx);
    out[bb * 10 + o] = logits[bb][o] - mx - logf(se);
  }
}

// ---------------- host orchestration ----------------
static inline unsigned gdim(long long n, int b) { return (unsigned)((n + b - 1) / b); }

extern "C" void kernel_launch(void* const* d_in, const int* in_sizes, int n_in,
                              void* d_out, int out_size, void* d_ws, size_t ws_size,
                              hipStream_t stream) {
  (void)in_sizes; (void)n_in; (void)out_size; (void)ws_size;
  const int N0 = 65536, N1 = 12288, N2 = 3072, N3 = 1024;
  const int E0 = 524288, E1 = 98304, E2 = 24576, E3 = 8192;

  const float* x0   = (const float*)d_in[0];
  const float* ea0  = (const float*)d_in[1];
  const float* pos0 = (const float*)d_in[2];
  const float* W_[4]  = {(const float*)d_in[3],  (const float*)d_in[8],  (const float*)d_in[13], (const float*)d_in[18]};
  const float* rt_[4] = {(const float*)d_in[4],  (const float*)d_in[9],  (const float*)d_in[14], (const float*)d_in[19]};
  const float* bi_[4] = {(const float*)d_in[5],  (const float*)d_in[10], (const float*)d_in[15], (const float*)d_in[20]};
  const float* ga_[4] = {(const float*)d_in[6],  (const float*)d_in[11], (const float*)d_in[16], (const float*)d_in[21]};
  const float* be_[4] = {(const float*)d_in[7],  (const float*)d_in[12], (const float*)d_in[17], (const float*)d_in[22]};
  const float* Wfc1 = (const float*)d_in[23];
  const float* bfc1 = (const float*)d_in[24];
  const float* Wfc2 = (const float*)d_in[25];
  const float* bfc2 = (const float*)d_in[26];
  const int* ei_[4] = {(const int*)d_in[27], (const int*)d_in[28], (const int*)d_in[29], (const int*)d_in[30]};
  const int* cl_[4] = {(const int*)d_in[31], (const int*)d_in[32], (const int*)d_in[33], (const int*)d_in[34]};

  // deterministic workspace carve
  char* cur = (char*)d_ws;
  auto carve = [&](size_t bytes) -> void* {
    void* p = (void*)cur;
    cur += (bytes + 255) & ~(size_t)255;
    return p;
  };
  float* maxabs = (float*)carve(4);
  float* pbuf   = (float*)carve(sizeof(float) * (size_t)E1 * 3);
  float* S      = (float*)carve(sizeof(float) * (size_t)N1 * NKER * 64);   // max across levels
  unsigned short* Wb = (unsigned short*)carve(sizeof(unsigned short) * (size_t)NKER * 256 * 512);
  float* deg    = (float*)carve(sizeof(float) * (size_t)N0);
  float* msg    = (float*)carve(sizeof(float) * (size_t)N0 * 64);
  float* h1     = (float*)carve(sizeof(float) * (size_t)N0 * 64);
  float* h1p    = (float*)carve(sizeof(float) * (size_t)N1 * 64);
  float* h2     = (float*)carve(sizeof(float) * (size_t)N1 * 128);
  float* h2p    = (float*)carve(sizeof(float) * (size_t)N2 * 128);
  float* h3     = (float*)carve(sizeof(float) * (size_t)N2 * 256);
  float* h3p    = (float*)carve(sizeof(float) * (size_t)N3 * 256);
  float* h4     = (float*)carve(sizeof(float) * (size_t)N3 * 512);
  float* hc     = (float*)carve(sizeof(float) * (size_t)512 * 512);
  float* pos1   = (float*)carve(sizeof(float) * (size_t)N1 * 3);
  float* pos2   = (float*)carve(sizeof(float) * (size_t)N2 * 3);
  float* pos3   = (float*)carve(sizeof(float) * (size_t)N3 * 3);
  float* cnt    = (float*)carve(sizeof(float) * (size_t)N1);
  float* psum   = (float*)carve(sizeof(float) * (size_t)N1 * 3);
  float* z1     = (float*)carve(sizeof(float) * (size_t)8 * 1024);

  auto run_level = [&](const float* x, int Nin, int cin, int cout, int E,
                       const int* ei, const float* pseudo, int lvl, float* hout) {
    int KK = NKER * cin;
    int KP = (KK + 63) & ~63;            // pad K to multiple of 64
    long long sN = (long long)Nin * KP;
    fill_f32<<<gdim(sN, 256), 256, 0, stream>>>(S, 0.f, sN);
    fill_f32<<<gdim(Nin, 256), 256, 0, stream>>>(deg, 0.f, Nin);
    spline_scatter<<<gdim((long long)E * 8, 256), 256, 0, stream>>>(x, ei, pseudo, S, deg, E, cin, KP);
    long long wN = (long long)KP * cout;
    cast_bf16_pad<<<gdim(wN, 256), 256, 0, stream>>>(W_[lvl], Wb, KK, KP, cout);
    dim3 gg(Nin / 32, cout / 64);
    wmma_gemm<<<gg, 256, 0, stream>>>(S, Wb, msg, Nin, cout, KP);
    epilogue_kernel<<<gdim((long long)Nin * cout, 256), 256, 0, stream>>>(
        msg, deg, x, rt_[lvl], bi_[lvl], hout, Nin, cin, cout);
    bn_kernel<<<cout, 256, 0, stream>>>(hout, ga_[lvl], be_[lvl], Nin, cout);
  };

  auto run_pool = [&](const float* hin, const float* posin, const int* cl, int Nin,
                      int nout, int cout, float* xm, float* posout) {
    fill_f32<<<gdim((long long)nout * cout, 256), 256, 0, stream>>>(xm, -FLT_MAX, (long long)nout * cout);
    fill_f32<<<gdim(nout, 256), 256, 0, stream>>>(cnt, 0.f, nout);
    int hasPos = posout != nullptr;
    if (hasPos) fill_f32<<<gdim((long long)nout * 3, 256), 256, 0, stream>>>(psum, 0.f, (long long)nout * 3);
    pool_scatter<<<gdim(Nin, 256), 256, 0, stream>>>(hin, posin, cl, xm, cnt, psum, Nin, cout, hasPos);
    pool_fin_x<<<gdim((long long)nout * cout, 256), 256, 0, stream>>>(xm, cnt, (long long)nout * cout, cout);
    if (hasPos) pool_fin_pos<<<gdim((long long)nout * 3, 256), 256, 0, stream>>>(psum, cnt, posout, nout);
  };

  auto make_pseudo = [&](const float* pos, const int* ei, int E) {
    fill_f32<<<1, 1, 0, stream>>>(maxabs, 0.f, 1);
    cart_max_kernel<<<gdim(E, 256), 256, 0, stream>>>(pos, ei, E, pbuf, maxabs);
    pseudo_kernel<<<gdim((long long)E * 3, 256), 256, 0, stream>>>(pbuf, maxabs, (long long)E * 3);
  };

  // level 1
  run_level(x0, N0, 1, 64, E0, ei_[0], ea0, 0, h1);
  run_pool(h1, pos0, cl_[0], N0, N1, 64, h1p, pos1);
  // level 2
  make_pseudo(pos1, ei_[1], E1);
  run_level(h1p, N1, 64, 128, E1, ei_[1], pbuf, 1, h2);
  run_pool(h2, pos1, cl_[1], N1, N2, 128, h2p, pos2);
  // level 3
  make_pseudo(pos2, ei_[2], E2);
  run_level(h2p, N2, 128, 256, E2, ei_[2], pbuf, 2, h3);
  run_pool(h3, pos2, cl_[2], N2, N3, 256, h3p, pos3);
  // level 4
  make_pseudo(pos3, ei_[3], E3);
  run_level(h3p, N3, 256, 512, E3, ei_[3], pbuf, 3, h4);
  // final cluster max-pool to [512, 512] == [8, 32768]
  run_pool(h4, nullptr, cl_[3], N3, 512, 512, hc, nullptr);
  // FC head
  fc1_kernel<<<gdim(8 * 1024, 256), 256, 0, stream>>>(hc, Wfc1, bfc1, z1, 64 * 512, 1024);
  fc2_lsm<<<1, 128, 0, stream>>>(z1, Wfc2, bfc2, (float*)d_out);
}
